// StatsSelect_85323820302343
// MI455X (gfx1250) — matrix operand order repaired and checked by reading, not compiled, not run on hardware
//
#include <hip/hip_runtime.h>
#include <hip/hip_bf16.h>
#include <math.h>

typedef __attribute__((ext_vector_type(4)))  float    v4f;
typedef __attribute__((ext_vector_type(8)))  float    v8f;
typedef __attribute__((ext_vector_type(8)))  _Float16 v8h;
typedef __attribute__((ext_vector_type(16))) _Float16 v16h;

#define B_   32
#define C_   512
#define T_   4000
#define T4_  1000          // T/4
#define HK_  2048          // 4*C
#define ROWS_ (B_*C_)      // 16384

// ---------------------------------------------------------------------------
// Pass 1: per-(b,c) raw power sums of x = x_a + x_b over T.  One block per row.
// 524 MB streamed with non-temporal b128 loads -> HBM-bound.
// ---------------------------------------------------------------------------
__global__ void moments_kernel(const float* __restrict__ xa,
                               const float* __restrict__ xb,
                               v4f* __restrict__ mom) {
  const int row = blockIdx.x;
  const int tid = threadIdx.x;
  const v4f* a4 = reinterpret_cast<const v4f*>(xa + (size_t)row * T_);
  const v4f* b4 = reinterpret_cast<const v4f*>(xb + (size_t)row * T_);

  float s1 = 0.f, s2 = 0.f, s3 = 0.f, s4 = 0.f;
  for (int i = tid; i < T4_; i += 256) {
    v4f x = __builtin_nontemporal_load(a4 + i) + __builtin_nontemporal_load(b4 + i);
#pragma unroll
    for (int j = 0; j < 4; ++j) {
      float v = x[j], v2 = v * v;
      s1 += v; s2 += v2; s3 += v2 * v; s4 += v2 * v2;
    }
  }

  __shared__ float red[4][256];
  red[0][tid] = s1; red[1][tid] = s2; red[2][tid] = s3; red[3][tid] = s4;
  __syncthreads();
  for (int off = 128; off > 0; off >>= 1) {
    if (tid < off) {
      red[0][tid] += red[0][tid + off];
      red[1][tid] += red[1][tid + off];
      red[2][tid] += red[2][tid + off];
      red[3][tid] += red[3][tid + off];
    }
    __syncthreads();
  }
  if (tid == 0) {
    v4f m = { red[0][0], red[1][0], red[2][0], red[3][0] };
    mom[row] = m;
  }
}

// ---------------------------------------------------------------------------
// Pass 2a: raw sums -> (mean, std_unbiased, skew, kurt), packed f16 rows of
// stats[B, 4C] laid out exactly like the reference concat.
// ---------------------------------------------------------------------------
__global__ void stats_kernel(const v4f* __restrict__ mom,
                             _Float16* __restrict__ stats) {
  const int idx = blockIdx.x * 256 + threadIdx.x;   // < ROWS_
  v4f m = mom[idx];
  const float T = (float)T_;
  const float mu  = m[0] / T;
  const float mu2 = mu * mu;
  const float M2 = m[1] - T * mu2;
  const float var = M2 / (T - 1.f);
  const float sd = sqrtf(fmaxf(var, 0.f));
  const float d  = fmaxf(sd, 0.01f);
  const float M3 = m[2] - 3.f * mu * m[1] + 2.f * T * mu * mu2;
  const float M4 = m[3] - 4.f * mu * m[2] + 6.f * mu2 * m[1] - 3.f * T * mu2 * mu2;
  const float d2 = d * d;
  const float skew = M3 / (T * d2 * d);
  const float kurt = M4 / (T * d2 * d2);

  const int b = idx >> 9, c = idx & (C_ - 1);
  _Float16* row = stats + b * HK_;
  row[c]            = (_Float16)mu;
  row[C_ + c]       = (_Float16)sd;
  row[2 * C_ + c]   = (_Float16)skew;
  row[3 * C_ + c]   = (_Float16)kurt;
}

// ---------------------------------------------------------------------------
// WMMA helpers: layouts used here for V_WMMA_F32_16X16X32_F16 (wave32); contiguous k, not ISA table order.
//   A (16x32): lane L holds row M=L&15; klo=(L>=16)?16:0, K = klo+0..15
//              contiguous  -> two contiguous v8h (16B) loads.
//   B (32x16): lane L holds col N=L&15; K = (L>=16?16:0)+0..15 contiguous.
//   D (16x16): VGPR r -> M = r + 8*(L>=16), N = L&15.
// ---------------------------------------------------------------------------
__device__ __forceinline__ v16h load_a_f16(const _Float16* __restrict__ arow,
                                           int kk, int kloA) {
  const v8h a0 = *reinterpret_cast<const v8h*>(arow + kk + kloA);
  const v8h a1 = *reinterpret_cast<const v8h*>(arow + kk + kloA + 8);
  return __builtin_shufflevector(a0, a1, 0,1,2,3,4,5,6,7,8,9,10,11,12,13,14,15);
}

// GEMM1: h[32,512] = stats[32,2048] @ W1^T + b1, output f16 for GEMM2.
// One wave per 16x16 tile; grid = (512/16, 32/16).
__global__ void gemm1_kernel(const _Float16* __restrict__ A,   // stats 32x2048 f16
                             const float* __restrict__ W1,     // 512x2048 f32
                             const float* __restrict__ b1,     // 512
                             _Float16* __restrict__ H) {       // 32x512 f16
  const int lane = threadIdx.x;
  const int n0 = blockIdx.x * 16;
  const int m0 = blockIdx.y * 16;
  const int half = lane >> 4;
  const int l15  = lane & 15;
  const int kloA = half * 16;
  const int kloB = half * 16;
  const _Float16* arow = A + (m0 + l15) * HK_;
  const float*    brow = W1 + (size_t)(n0 + l15) * HK_;

  v8f acc = {};
  for (int kk = 0; kk < HK_; kk += 32) {
    v16h av = load_a_f16(arow, kk, kloA);
    const v4f* bp = reinterpret_cast<const v4f*>(brow + kk + kloB);
    v4f f0 = bp[0], f1 = bp[1], f2 = bp[2], f3 = bp[3];
    v16h bv;
#pragma unroll
    for (int j = 0; j < 4; ++j) {
      bv[j]      = (_Float16)f0[j];
      bv[4 + j]  = (_Float16)f1[j];
      bv[8 + j]  = (_Float16)f2[j];
      bv[12 + j] = (_Float16)f3[j];
    }
    acc = __builtin_amdgcn_wmma_f32_16x16x32_f16(false, av, false, bv,
                                                 (short)0, acc, false, false);
  }
  const float bias = b1[n0 + l15];
#pragma unroll
  for (int r = 0; r < 8; ++r) {
    const int mrow = m0 + r + half * 8;
    H[mrow * C_ + n0 + l15] = (_Float16)(acc[r] + bias);
  }
}

// GEMM2 + softmax-over-2: g = sigmoid(h @ (W2[0]-W2[1])^T + (b2[0]-b2[1])).
__global__ void gemm2_kernel(const _Float16* __restrict__ H,   // 32x512 f16
                             const float* __restrict__ W2,     // 2x512x512 f32
                             const float* __restrict__ b2,     // 2x512 f32
                             float* __restrict__ G) {          // 32x512 f32 gate
  const int lane = threadIdx.x;
  const int n0 = blockIdx.x * 16;
  const int m0 = blockIdx.y * 16;
  const int half = lane >> 4;
  const int l15  = lane & 15;
  const int kloA = half * 16;
  const int kloB = half * 16;
  const _Float16* arow = H + (m0 + l15) * C_;
  const float* w0 = W2 + (size_t)(n0 + l15) * C_;
  const float* w1 = w0 + (size_t)C_ * C_;

  v8f acc = {};
  for (int kk = 0; kk < C_; kk += 32) {
    v16h av = load_a_f16(arow, kk, kloA);
    const v4f* p0 = reinterpret_cast<const v4f*>(w0 + kk + kloB);
    const v4f* p1 = reinterpret_cast<const v4f*>(w1 + kk + kloB);
    v16h bv;
#pragma unroll
    for (int q = 0; q < 4; ++q) {
      v4f d = p0[q] - p1[q];
#pragma unroll
      for (int j = 0; j < 4; ++j) bv[q * 4 + j] = (_Float16)d[j];
    }
    acc = __builtin_amdgcn_wmma_f32_16x16x32_f16(false, av, false, bv,
                                                 (short)0, acc, false, false);
  }
  const float bias = b2[n0 + l15] - b2[C_ + n0 + l15];
#pragma unroll
  for (int r = 0; r < 8; ++r) {
    const int mrow = m0 + r + half * 8;
    const float s = acc[r] + bias;
    G[mrow * C_ + n0 + l15] = 1.f / (1.f + expf(-s));  // softmax_k=2 branch-0 weight
  }
}

// ---------------------------------------------------------------------------
// Pass 3: out = g*x_a + (1-g)*x_b.  786 MB streamed non-temporally.
// ---------------------------------------------------------------------------
__global__ void blend_kernel(const float* __restrict__ xa,
                             const float* __restrict__ xb,
                             const float* __restrict__ G,
                             float* __restrict__ out) {
  const int row = blockIdx.x;
  const float g  = G[row];
  const float g1 = 1.f - g;
  const v4f* a4 = reinterpret_cast<const v4f*>(xa + (size_t)row * T_);
  const v4f* b4 = reinterpret_cast<const v4f*>(xb + (size_t)row * T_);
  v4f* o4 = reinterpret_cast<v4f*>(out + (size_t)row * T_);
  for (int i = threadIdx.x; i < T4_; i += 256) {
    v4f a = __builtin_nontemporal_load(a4 + i);
    v4f b = __builtin_nontemporal_load(b4 + i);
    v4f o = a * g + b * g1;
    __builtin_nontemporal_store(o, o4 + i);
  }
}

// ---------------------------------------------------------------------------
extern "C" void kernel_launch(void* const* d_in, const int* in_sizes, int n_in,
                              void* d_out, int out_size, void* d_ws, size_t ws_size,
                              hipStream_t stream) {
  (void)in_sizes; (void)n_in; (void)out_size; (void)ws_size;
  const float* xa = (const float*)d_in[0];
  const float* xb = (const float*)d_in[1];
  const float* W1 = (const float*)d_in[2];
  const float* b1 = (const float*)d_in[3];
  const float* W2 = (const float*)d_in[4];
  const float* b2 = (const float*)d_in[5];
  float* out = (float*)d_out;

  char* ws = (char*)d_ws;
  v4f*      mom   = (v4f*)ws;                                  // 16384*16  = 256 KB
  _Float16* stats = (_Float16*)(ws + 262144);                  // 32*2048*2 = 128 KB
  _Float16* H     = (_Float16*)(ws + 262144 + 131072);         // 32*512*2  =  32 KB
  float*    G     = (float*)(ws + 262144 + 131072 + 32768);    // 32*512*4  =  64 KB

  moments_kernel<<<ROWS_, 256, 0, stream>>>(xa, xb, mom);
  stats_kernel<<<ROWS_ / 256, 256, 0, stream>>>(mom, stats);
  gemm1_kernel<<<dim3(C_ / 16, B_ / 16), 32, 0, stream>>>(stats, W1, b1, H);
  gemm2_kernel<<<dim3(C_ / 16, B_ / 16), 32, 0, stream>>>(H, W2, b2, G);
  blend_kernel<<<ROWS_, 256, 0, stream>>>(xa, xb, G, out);
}
